// attention_unet_13950053777633
// MI455X (gfx1250) — matrix operand order repaired and checked by reading, not compiled, not run on hardware
//
#include <hip/hip_runtime.h>
#include <math.h>

// ---------------------------------------------------------------------------
// Attention-UNet forward for MI455X (gfx1250, wave32, WMMA).
// Convs run as implicit GEMM on v_wmma_f32_16x16x32_bf16 (bf16 in, fp32 acc).
// Main conv path: LDS-staged, tap-major K ordering, 4 waves/block sharing the
// staged activation slab (64 outch x 16 pixels per block). Fallback
// direct-gather path only for CI=4 (inc conv1) and the 8x8 level (down4).
// All spatial/channel dims are powers of two -> shift/mask addressing, image
// index on blockIdx.z (no integer division anywhere on the hot path).
// Elementwise stages (BN, blur-down, bilinear, T=8 attention core, 1x1 head)
// are bandwidth-bound fp32 VALU. Skip-concats are fused into the conv input
// read (split-channel) to avoid materializing concat buffers.
// ---------------------------------------------------------------------------

typedef __attribute__((ext_vector_type(16))) __bf16 v16bf;
typedef __attribute__((ext_vector_type(8)))  __bf16 v8bf;
typedef __attribute__((ext_vector_type(8)))  float  v8f;

__device__ __forceinline__ int iclamp(int v, int lo, int hi) {
    return v < lo ? lo : (v > hi ? hi : v);
}

static inline int ilog2(int v) { int s = 0; while ((1 << s) < v) ++s; return s; }

// ---------------------------------------------------------------------------
// Weight prepack: fp32 (CO,CI,KS,KS) -> bf16 [CO][Kpad], zero padded in K.
// tapMajor=0: k' = ci*KS*KS + tap   (direct-gather kernel order)
// tapMajor=1: k' = tap*CI + ci      (LDS-staged kernel order)
// Grid: x covers Kpad, y = co. CI is a power of two (CIsh = log2 CI).
// ---------------------------------------------------------------------------
__global__ __launch_bounds__(256) void pack_w_kernel(
    const float* __restrict__ w, __bf16* __restrict__ wp,
    int CIsh, int KSS, int Kpad, int tapMajor)
{
    const int k  = blockIdx.x * 256 + threadIdx.x;
    if (k >= Kpad) return;
    const int co = blockIdx.y;
    const int CI = 1 << CIsh;
    const int K  = CI * KSS;
    float v = 0.0f;
    if (k < K) {
        int srck;
        if (tapMajor) {
            int tap = k >> CIsh;
            int ci  = k & (CI - 1);
            srck = ci * KSS + tap;
        } else {
            srck = k;
        }
        v = w[(size_t)co * K + srck];
    }
    wp[(size_t)co * Kpad + k] = (__bf16)v;
}

// ---------------------------------------------------------------------------
// LDS-staged implicit-GEMM conv. Requires W%16==0, CI%32==0, csplit%32==0,
// CO%64==0. Block = 128 threads (4 waves); each wave owns a 16-outch tile,
// all share one 16-pixel row tile. K order is tap-major: each 32-wide WMMA
// chunk = one (kh,kw) tap x 32 channels. Per (ci-chunk, kh) we stage a
// 32ch x 18col fp32 slab once and consume it for kw = 0,1,2.
// A (16x32 bf16): lane l = row, half-wave h covers K h*16..h*16+15.
// B (32x16 bf16): lane l = column, half-wave h covers K h*16..h*16+15.
// D (16x16 f32): VGPR v -> M = h*8+v, N = lane&15.
// Grid: x = pixel tile in image, y = co block, z = image.
// ---------------------------------------------------------------------------
template<int KS>
__global__ __launch_bounds__(128) void conv_wmma_lds_kernel(
    const float* __restrict__ xa, const float* __restrict__ xb, int csplit,
    int CI, int Wsh, int HWsh,
    const __bf16* __restrict__ wp, int Kpad,
    const float* __restrict__ bias,
    float* __restrict__ y, int CO)
{
    constexpr int NTAP = (KS == 3) ? 3 : 1;   // taps per axis
    constexpr int HALO = (KS == 3) ? 1 : 0;
    constexpr int COLS = 16 + 2 * HALO;       // staged columns
    constexpr int PITCH = 19;                 // bank-conflict-free pitch

    __shared__ float smem[32 * PITCH];

    const int W = 1 << Wsh;
    const int H = 1 << (HWsh - Wsh);

    const int tid  = threadIdx.x;
    const int wave = tid >> 5;
    const int lane = tid & 31;
    const int half = lane >> 4;
    const int l    = lane & 15;

    const int n    = blockIdx.z;
    const int pix0 = blockIdx.x << 4;         // 16 pixels, same image row
    const int h    = pix0 >> Wsh;
    const int w0   = pix0 & (W - 1);          // multiple of 16

    const int coBase = blockIdx.y * 64 + wave * 16;
    const int m      = coBase + l;
    const __bf16* wrow = wp + (size_t)m * Kpad;

    v8f acc = {};

    for (int ci0 = 0; ci0 < CI; ci0 += 32) {
        // concat-split: whole 32-channel chunk lives in xa or xb
        const float* src;
        int cbase, cdim;
        if (ci0 < csplit) { src = xa; cbase = ci0;          cdim = csplit;      }
        else              { src = xb; cbase = ci0 - csplit; cdim = CI - csplit; }
        const size_t chanBase = (size_t)(n * cdim + cbase) << HWsh;

        #pragma unroll
        for (int dh = 0; dh < NTAP; ++dh) {
            const int ih = h + dh - HALO;

            __syncthreads();
            // cooperative stage: 32 channels x COLS columns (zero-fill OOB)
            for (int e = tid; e < 32 * COLS; e += 128) {
                const int c   = e / COLS;            // COLS=18 or 16, unrolled small
                const int col = e - c * COLS;
                const int iw  = w0 + col - HALO;
                float v = 0.0f;
                if (ih >= 0 && ih < H && iw >= 0 && iw < W)
                    v = src[chanBase + ((size_t)c << HWsh) + (ih << Wsh) + iw];
                smem[c * PITCH + col] = v;
            }
            __syncthreads();

            #pragma unroll
            for (int kw = 0; kw < NTAP; ++kw) {
                const int kc = (KS == 3 ? (dh * 3 + kw) * CI : 0) + ci0;

                v8bf a0 = *(const v8bf*)(wrow + kc + half * 16);
                v8bf a1 = *(const v8bf*)(wrow + kc + half * 16 + 8);
                v16bf A = __builtin_shufflevector(a0, a1,
                            0,1,2,3,4,5,6,7,8,9,10,11,12,13,14,15);

                v16bf Bv = {};
                #pragma unroll
                for (int j = 0; j < 16; ++j)
                    Bv[j] = (__bf16)smem[(half * 16 + j) * PITCH + l + kw];

                acc = __builtin_amdgcn_wmma_f32_16x16x32_bf16(
                          false, A, false, Bv, (short)0, acc, false, false);
            }
        }
    }

    #pragma unroll
    for (int v = 0; v < 8; ++v) {
        const int co = coBase + half * 8 + v;
        float o = acc[v] + (bias ? bias[co] : 0.0f);
        y[((size_t)(n * CO + co) << HWsh) + pix0 + l] = o;
    }
}

// ---------------------------------------------------------------------------
// Direct-gather fallback conv (any W / CI). One wave per block, 16x16 tile.
// Used only for inc conv1 (CI=4) and the 8x8 level (down4): <5% of FLOPs.
// K order: ci-major (k = ci*9 + tap). Grid: x = pixel tile, y = co, z = image.
// ---------------------------------------------------------------------------
template<int KS>
__global__ __launch_bounds__(32) void conv_wmma_kernel(
    const float* __restrict__ xa, const float* __restrict__ xb, int csplit,
    int CI, int Wsh, int HWsh,
    const __bf16* __restrict__ wp, int Kpad, int Ktot,
    const float* __restrict__ bias,
    float* __restrict__ y, int CO)
{
    const int W = 1 << Wsh;
    const int H = 1 << (HWsh - Wsh);

    const int lane = threadIdx.x & 31;
    const int half = lane >> 4;
    const int l    = lane & 15;

    const int n   = blockIdx.z;
    const int pix = (blockIdx.x << 4) + l;
    const int h   = pix >> Wsh;
    const int w   = pix & (W - 1);

    const int coBase = blockIdx.y * 16;
    const int m      = coBase + l;
    const __bf16* wrow = wp + (size_t)m * Kpad;

    v8f acc = {};

    for (int kc = 0; kc < Ktot; kc += 32) {
        v8bf a0 = *(const v8bf*)(wrow + kc + half * 16);
        v8bf a1 = *(const v8bf*)(wrow + kc + half * 16 + 8);
        v16bf A = __builtin_shufflevector(a0, a1,
                    0,1,2,3,4,5,6,7,8,9,10,11,12,13,14,15);

        v16bf Bv = {};
        const int kb = kc + half * 16;
        #pragma unroll
        for (int j = 0; j < 16; ++j) {
            int k = kb + j;
            float xv = 0.0f;
            if (k < Ktot) {
                int ci, ih, iw;
                if (KS == 3) {
                    ci = k / 9;             // compile-time constant divisor
                    int r  = k - ci * 9;
                    int kh = r / 3;
                    int kw = r - kh * 3;
                    ih = h + kh - 1;
                    iw = w + kw - 1;
                } else {
                    ci = k; ih = h; iw = w;
                }
                if (ih >= 0 && ih < H && iw >= 0 && iw < W) {
                    const float* src;
                    int cc, cdim;
                    if (ci < csplit) { src = xa; cc = ci;          cdim = csplit;      }
                    else             { src = xb; cc = ci - csplit; cdim = CI - csplit; }
                    xv = src[((size_t)(n * cdim + cc) << HWsh) + (ih << Wsh) + iw];
                }
            }
            Bv[j] = (__bf16)xv;
        }

        acc = __builtin_amdgcn_wmma_f32_16x16x32_bf16(
                  false, A, false, Bv, (short)0, acc, false, false);
    }

    #pragma unroll
    for (int v = 0; v < 8; ++v) {
        int co = coBase + half * 8 + v;
        float o = acc[v] + (bias ? bias[co] : 0.0f);
        y[((size_t)(n * CO + co) << HWsh) + pix] = o;
    }
}

// ---------------------------------------------------------------------------
// Training-mode BatchNorm: per-channel mean / rsqrt(var+eps) over (N,H,W).
// ---------------------------------------------------------------------------
__global__ __launch_bounds__(256) void bn_stats_kernel(
    const float* __restrict__ y, int Nimg, int C, int HWsh,
    float* __restrict__ mean, float* __restrict__ rstd)
{
    const int c  = blockIdx.x;
    const int HW = 1 << HWsh;
    __shared__ float s1[256];
    __shared__ float s2[256];
    float a = 0.0f, b = 0.0f;
    const long total = (long)Nimg << HWsh;
    for (long i = threadIdx.x; i < total; i += 256) {
        long n = i >> HWsh;
        long r = i & (HW - 1);
        float v = y[((size_t)(n * C + c) << HWsh) + r];
        a += v;
        b += v * v;
    }
    s1[threadIdx.x] = a;
    s2[threadIdx.x] = b;
    __syncthreads();
    for (int st = 128; st > 0; st >>= 1) {
        if (threadIdx.x < st) {
            s1[threadIdx.x] += s1[threadIdx.x + st];
            s2[threadIdx.x] += s2[threadIdx.x + st];
        }
        __syncthreads();
    }
    if (threadIdx.x == 0) {
        float mm  = s1[0] / (float)total;
        float var = s2[0] / (float)total - mm * mm;
        mean[c] = mm;
        rstd[c] = rsqrtf(var + 1e-5f);
    }
}

__global__ __launch_bounds__(256) void bn_apply_kernel(
    float* __restrict__ y, const float* __restrict__ mean,
    const float* __restrict__ rstd, const float* __restrict__ g,
    const float* __restrict__ be, int Cmask, int HWsh, long total)
{
    long i = (long)blockIdx.x * 256 + threadIdx.x;
    if (i >= total) return;
    int c = (int)((i >> HWsh) & Cmask);
    float v = (y[i] - mean[c]) * rstd[c] * g[c] + be[c];
    y[i] = v > 0.0f ? v : 0.0f;
}

// ---------------------------------------------------------------------------
// Per-pixel temporal attention core (T = 8 fixed).
// att[p][q] = sum_c Q[p,c]K[q,c]; softmax over p; out[q,c] = sum_p att*V[p,c].
// ---------------------------------------------------------------------------
__global__ __launch_bounds__(256) void attn_kernel(
    const float* __restrict__ Q, const float* __restrict__ Kt,
    const float* __restrict__ V, float* __restrict__ out,
    int Bb, int C, int HWsh)
{
    const int HW = 1 << HWsh;
    const long total = ((long)Bb * 8) << HWsh;
    long i = (long)blockIdx.x * 256 + threadIdx.x;
    if (i >= total) return;
    const int pix = (int)(i & (HW - 1));
    const int q   = (int)((i >> HWsh) & 7);
    const int b   = (int)(i >> (HWsh + 3));

    float att[8];
    float mx = -3.4e38f;
    #pragma unroll
    for (int p = 0; p < 8; ++p) {
        const float* qp = Q + ((size_t)((b * 8 + p) * C) << HWsh) + pix;
        const float* kp = Kt + ((size_t)((b * 8 + q) * C) << HWsh) + pix;
        float d = 0.0f;
        for (int c = 0; c < C; ++c)
            d += qp[(size_t)c << HWsh] * kp[(size_t)c << HWsh];
        att[p] = d;
        mx = fmaxf(mx, d);
    }
    float s = 0.0f;
    #pragma unroll
    for (int p = 0; p < 8; ++p) { att[p] = __expf(att[p] - mx); s += att[p]; }
    const float inv = 1.0f / s;
    #pragma unroll
    for (int p = 0; p < 8; ++p) att[p] *= inv;

    for (int c = 0; c < C; ++c) {
        float o = 0.0f;
        #pragma unroll
        for (int p = 0; p < 8; ++p)
            o += att[p] * V[((size_t)((b * 8 + p) * C + c) << HWsh) + pix];
        out[((size_t)((b * 8 + q) * C + c) << HWsh) + pix] = o;
    }
}

// ---------------------------------------------------------------------------
// Fused anti-aliased downsample: maxpool(2,1,VALID) -> reflect pad 1 ->
// 3x3 binomial / 16, stride 2. H == W (square), OWsh = log2(W/2).
// ---------------------------------------------------------------------------
__global__ __launch_bounds__(256) void blur_down_kernel(
    const float* __restrict__ x, float* __restrict__ y,
    int NC, int OWsh)
{
    const int OW = 1 << OWsh;
    const int W  = OW * 2, H = W;
    const long total = ((long)NC) << (2 * OWsh);
    long i = (long)blockIdx.x * 256 + threadIdx.x;
    if (i >= total) return;
    const int ow = (int)(i & (OW - 1));
    const int oh = (int)((i >> OWsh) & (OW - 1));
    const int nc = (int)(i >> (2 * OWsh));
    const float* xp = x + ((size_t)nc * H) * W;
    const int Hm = H - 1, Wm = W - 1;        // maxpool output dims
    const float kf[3] = {1.0f, 2.0f, 1.0f};
    float acc = 0.0f;
    #pragma unroll
    for (int a = 0; a < 3; ++a) {
        int r = 2 * oh - 1 + a;
        if (r < 0) r = -r;
        if (r >= Hm) r = 2 * Hm - 2 - r;
        #pragma unroll
        for (int bj = 0; bj < 3; ++bj) {
            int c = 2 * ow - 1 + bj;
            if (c < 0) c = -c;
            if (c >= Wm) c = 2 * Wm - 2 - c;
            float m0 = fmaxf(xp[(size_t)r * W + c], xp[(size_t)r * W + c + 1]);
            float m1 = fmaxf(xp[(size_t)(r + 1) * W + c], xp[(size_t)(r + 1) * W + c + 1]);
            acc += kf[a] * kf[bj] * fmaxf(m0, m1);
        }
    }
    y[i] = acc * (1.0f / 16.0f);
}

// ---------------------------------------------------------------------------
// 2x bilinear upsample, half-pixel centers, edge clamp. Square, Wsh = log2 W.
// ---------------------------------------------------------------------------
__global__ __launch_bounds__(256) void upsample2x_kernel(
    const float* __restrict__ x, float* __restrict__ y,
    int NC, int Wsh)
{
    const int W  = 1 << Wsh, H = W;
    const int OW = W * 2;
    const int OWsh = Wsh + 1;
    const long total = ((long)NC) << (2 * OWsh);
    long i = (long)blockIdx.x * 256 + threadIdx.x;
    if (i >= total) return;
    const int ow = (int)(i & (OW - 1));
    const int oh = (int)((i >> OWsh) & (OW - 1));
    const int nc = (int)(i >> (2 * OWsh));
    const float* xp = x + ((size_t)nc * H) * W;

    float sy = (oh + 0.5f) * 0.5f - 0.5f;
    float sx = (ow + 0.5f) * 0.5f - 0.5f;
    int y0 = (int)floorf(sy);
    int x0 = (int)floorf(sx);
    float fy = sy - y0;
    float fx = sx - x0;
    int y0c = iclamp(y0, 0, H - 1), y1c = iclamp(y0 + 1, 0, H - 1);
    int x0c = iclamp(x0, 0, W - 1), x1c = iclamp(x0 + 1, 0, W - 1);
    float v = (1.0f - fy) * ((1.0f - fx) * xp[(size_t)y0c * W + x0c] +
                             fx          * xp[(size_t)y0c * W + x1c]) +
              fy          * ((1.0f - fx) * xp[(size_t)y1c * W + x0c] +
                             fx          * xp[(size_t)y1c * W + x1c]);
    y[i] = v;
}

// ---------------------------------------------------------------------------
// Final 1x1 conv head (CO = 2: too small for WMMA, bandwidth bound anyway).
// ---------------------------------------------------------------------------
__global__ __launch_bounds__(256) void conv1x1_out_kernel(
    const float* __restrict__ x, const float* __restrict__ w,
    const float* __restrict__ bias, float* __restrict__ y,
    int Nimg, int CI, int COsh, int HWsh)
{
    const int HW = 1 << HWsh;
    const int CO = 1 << COsh;
    const long total = ((long)Nimg) << (HWsh + COsh);
    long i = (long)blockIdx.x * 256 + threadIdx.x;
    if (i >= total) return;
    const int pix = (int)(i & (HW - 1));
    const int co  = (int)((i >> HWsh) & (CO - 1));
    const int n   = (int)(i >> (HWsh + COsh));
    const float* xp = x + ((size_t)(n * CI) << HWsh) + pix;
    float acc = bias[co];
    for (int c = 0; c < CI; ++c)
        acc += w[co * CI + c] * xp[(size_t)c << HWsh];
    y[i] = acc;
}

// ---------------------------------------------------------------------------
// Host-side orchestration
// ---------------------------------------------------------------------------
static void run_conv(hipStream_t s, const float* xa, const float* xb, int csplit,
                     int Nimg, int CI, int H, int W,
                     const float* w, const float* bias,
                     float* y, int CO, int KS, __bf16* wpack)
{
    const int KSS  = KS * KS;
    const int K    = CI * KSS;
    const int Kpad = (K + 31) & ~31;
    const int Wsh  = ilog2(W);
    const int HWsh = ilog2(H * W);

    const bool staged = (W % 16 == 0) && (CI % 32 == 0) &&
                        (csplit % 32 == 0) && (CO % 64 == 0);

    pack_w_kernel<<<dim3((unsigned)((Kpad + 255) / 256), (unsigned)CO),
                    dim3(256), 0, s>>>(w, wpack, ilog2(CI), KSS, Kpad,
                                       staged ? 1 : 0);

    if (staged) {
        dim3 grid((unsigned)((H * W) / 16), (unsigned)(CO / 64), (unsigned)Nimg);
        if (KS == 3)
            conv_wmma_lds_kernel<3><<<grid, dim3(128), 0, s>>>(
                xa, xb, csplit, CI, Wsh, HWsh, wpack, Kpad, bias, y, CO);
        else
            conv_wmma_lds_kernel<1><<<grid, dim3(128), 0, s>>>(
                xa, xb, csplit, CI, Wsh, HWsh, wpack, Kpad, bias, y, CO);
    } else {
        dim3 grid((unsigned)((H * W) / 16), (unsigned)(CO / 16), (unsigned)Nimg);
        if (KS == 3)
            conv_wmma_kernel<3><<<grid, dim3(32), 0, s>>>(
                xa, xb, csplit, CI, Wsh, HWsh, wpack, Kpad, K, bias, y, CO);
        else
            conv_wmma_kernel<1><<<grid, dim3(32), 0, s>>>(
                xa, xb, csplit, CI, Wsh, HWsh, wpack, Kpad, K, bias, y, CO);
    }
}

static void run_bn(hipStream_t s, float* y, int Nimg, int C, int HW,
                   const float* g, const float* be, float* mean, float* rstd)
{
    const int HWsh = ilog2(HW);
    bn_stats_kernel<<<dim3((unsigned)C), dim3(256), 0, s>>>(
        y, Nimg, C, HWsh, mean, rstd);
    const long total = (long)Nimg * C * HW;
    bn_apply_kernel<<<dim3((unsigned)((total + 255) / 256)), dim3(256), 0, s>>>(
        y, mean, rstd, g, be, C - 1, HWsh, total);
}

// _dcp fields (sorted-key flatten): +0 b1 +1 b2 +2 be1 +3 be2 +4 g1 +5 g2 +6 w1 +7 w2
static void run_dconv(hipStream_t s, void* const* d_in, int base,
                      const float* xa, const float* xb, int csplit,
                      int Nimg, int CI, int H, int W, int CO,
                      float* tmp, float* out, __bf16* wpack,
                      float* mean, float* rstd)
{
    const float* be1 = (const float*)d_in[base + 2];
    const float* be2 = (const float*)d_in[base + 3];
    const float* g1  = (const float*)d_in[base + 4];
    const float* g2  = (const float*)d_in[base + 5];
    const float* w1  = (const float*)d_in[base + 6];
    const float* w2  = (const float*)d_in[base + 7];
    // conv bias before training-mode BN cancels in the mean -> skipped.
    run_conv(s, xa, xb, csplit, Nimg, CI, H, W, w1, nullptr, tmp, CO, 3, wpack);
    run_bn(s, tmp, Nimg, CO, H * W, g1, be1, mean, rstd);
    run_conv(s, tmp, nullptr, CO, Nimg, CO, H, W, w2, nullptr, out, CO, 3, wpack);
    run_bn(s, out, Nimg, CO, H * W, g2, be2, mean, rstd);
}

// _attp fields (sorted-key flatten): +0 bk +1 bq +2 bv +3 wk +4 wq +5 wv
static void run_attn(hipStream_t s, void* const* d_in, int base, float* x,
                     int Bb, int C, int H, int W,
                     float* q, float* k, float* v, __bf16* wpack)
{
    const int Nimg = Bb * 8;
    const int HWsh = ilog2(H * W);
    const float* bk = (const float*)d_in[base + 0];
    const float* bq = (const float*)d_in[base + 1];
    const float* bv = (const float*)d_in[base + 2];
    const float* wk = (const float*)d_in[base + 3];
    const float* wq = (const float*)d_in[base + 4];
    const float* wv = (const float*)d_in[base + 5];
    run_conv(s, x, nullptr, C, Nimg, C, H, W, wq, bq, q, C, 1, wpack);
    run_conv(s, x, nullptr, C, Nimg, C, H, W, wk, bk, k, C, 1, wpack);
    run_conv(s, x, nullptr, C, Nimg, C, H, W, wv, bv, v, C, 1, wpack);
    const long total = ((long)Bb * 8) << HWsh;
    attn_kernel<<<dim3((unsigned)((total + 255) / 256)), dim3(256), 0, s>>>(
        q, k, v, x, Bb, C, HWsh);
}

extern "C" void kernel_launch(void* const* d_in, const int* in_sizes, int n_in,
                              void* d_out, int out_size, void* d_ws, size_t ws_size,
                              hipStream_t stream)
{
    (void)in_sizes; (void)n_in; (void)out_size; (void)ws_size;

    // jax.tree (sorted dict keys at every level) flatten of setup_inputs():
    // 0 images
    // att0: 1..6   att1: 7..12
    // down1: 13..20  down2: 21..28  down3: 29..36  down4: 37..44
    // inc: 45..52   outc: 53 b, 54 w
    // up1: 55..62   up2: 63..70   up3: 71..78   up4: 79..86
    const float* images = (const float*)d_in[0];

    const int Bb = 2, Nimg = 16, CIN = 4;

    // Workspace layout (floats). Peak ~318.5 MiB.
    float* ws   = (float*)d_ws;
    float* x1a  = ws;                  // 16,777,216  (16,64,128,128)
    float* x2a  = ws + 16777216;       //  8,388,608  (16,128,64,64)
    float* x3   = ws + 25165824;       //  4,194,304  (16,256,32,32)
    float* x4   = ws + 29360128;       //  2,097,152  (16,512,16,16)
    float* x5   = ws + 31457280;       //    524,288  (16,512,8,8)
    float* tA   = ws + 31981568;       // 16,777,216 ping
    float* tB   = ws + 48758784;       // 16,777,216 pong
    float* tC   = ws + 65536000;       // 16,777,216 (QKV third / upsample)
    __bf16* wpack = (__bf16*)(ws + 82313216); // 2,359,296 bf16 (max CO*Kpad)
    float* mean = ws + 83492864;       // 1024
    float* rstd = ws + 83493888;       // 1024

    hipStream_t s = stream;

    // ---- encoder ----
    run_dconv(s, d_in, 45, images, nullptr, CIN, Nimg, CIN, 128, 128, 64,
              tA, x1a, wpack, mean, rstd);                       // inc (direct: CI=4)
    run_attn(s, d_in, 1, x1a, Bb, 64, 128, 128, tA, tB, tC, wpack); // att0

    {
        long t = (long)Nimg * 64 * 64 * 64;
        blur_down_kernel<<<dim3((unsigned)((t + 255) / 256)), dim3(256), 0, s>>>(
            x1a, tA, Nimg * 64, 6);
    }
    run_dconv(s, d_in, 13, tA, nullptr, 64, Nimg, 64, 64, 64, 128,
              tB, x2a, wpack, mean, rstd);                       // down1
    run_attn(s, d_in, 7, x2a, Bb, 128, 64, 64, tA, tB, tC, wpack);  // att1

    {
        long t = (long)Nimg * 128 * 32 * 32;
        blur_down_kernel<<<dim3((unsigned)((t + 255) / 256)), dim3(256), 0, s>>>(
            x2a, tA, Nimg * 128, 5);
    }
    run_dconv(s, d_in, 21, tA, nullptr, 128, Nimg, 128, 32, 32, 256,
              tB, x3, wpack, mean, rstd);                        // down2

    {
        long t = (long)Nimg * 256 * 16 * 16;
        blur_down_kernel<<<dim3((unsigned)((t + 255) / 256)), dim3(256), 0, s>>>(
            x3, tA, Nimg * 256, 4);
    }
    run_dconv(s, d_in, 29, tA, nullptr, 256, Nimg, 256, 16, 16, 512,
              tB, x4, wpack, mean, rstd);                        // down3

    {
        long t = (long)Nimg * 512 * 8 * 8;
        blur_down_kernel<<<dim3((unsigned)((t + 255) / 256)), dim3(256), 0, s>>>(
            x4, tA, Nimg * 512, 3);
    }
    run_dconv(s, d_in, 37, tA, nullptr, 512, Nimg, 512, 8, 8, 512,
              tB, x5, wpack, mean, rstd);                        // down4 (direct: W=8)

    // ---- decoder (skip-concat fused into conv input read) ----
    {
        long t = (long)Nimg * 512 * 16 * 16;
        upsample2x_kernel<<<dim3((unsigned)((t + 255) / 256)), dim3(256), 0, s>>>(
            x5, tC, Nimg * 512, 3);
    }
    run_dconv(s, d_in, 55, x4, tC, 512, Nimg, 1024, 16, 16, 256,
              tA, tB, wpack, mean, rstd);                        // up1

    {
        long t = (long)Nimg * 256 * 32 * 32;
        upsample2x_kernel<<<dim3((unsigned)((t + 255) / 256)), dim3(256), 0, s>>>(
            tB, tC, Nimg * 256, 4);
    }
    run_dconv(s, d_in, 63, x3, tC, 256, Nimg, 512, 32, 32, 128,
              tA, tB, wpack, mean, rstd);                        // up2

    {
        long t = (long)Nimg * 128 * 64 * 64;
        upsample2x_kernel<<<dim3((unsigned)((t + 255) / 256)), dim3(256), 0, s>>>(
            tB, tC, Nimg * 128, 5);
    }
    run_dconv(s, d_in, 71, x2a, tC, 128, Nimg, 256, 64, 64, 64,
              tA, tB, wpack, mean, rstd);                        // up3

    {
        long t = (long)Nimg * 64 * 128 * 128;
        upsample2x_kernel<<<dim3((unsigned)((t + 255) / 256)), dim3(256), 0, s>>>(
            tB, tC, Nimg * 64, 6);
    }
    run_dconv(s, d_in, 79, x1a, tC, 64, Nimg, 128, 128, 128, 64,
              tA, tB, wpack, mean, rstd);                        // up4

    // ---- 1x1 head -> d_out (2,8,2,128,128) fp32 ----
    {
        const float* ob = (const float*)d_in[53];
        const float* ow = (const float*)d_in[54];
        long t = (long)Nimg * 2 * 128 * 128;
        conv1x1_out_kernel<<<dim3((unsigned)((t + 255) / 256)), dim3(256), 0, s>>>(
            tB, ow, ob, (float*)d_out, Nimg, 64, 1, 14);
    }
}
